// HF_TransformerEncoderLayer_25254407700892
// MI455X (gfx1250) — compile-verified, matrix operand order checked by reading, not run on hardware
//
#include <hip/hip_runtime.h>
#include <hip/hip_bf16.h>
#include <math.h>
#include <stdint.h>

// ---------------- problem constants ----------------
static constexpr int B = 4, S = 2048, D = 512, H = 8, KD = 64, FFN = 2048;
static constexpr int NTOK = B * S;

// Async global->LDS fills for the GEMM mainloop (ASYNCcnt path).
// Set to 0 to fall back to VGPR-staged synchronous copies.
#define USE_ASYNC_CP 1

typedef _Float16 f16;
typedef __attribute__((ext_vector_type(16))) _Float16 v16h;
typedef __attribute__((ext_vector_type(8)))  _Float16 v8h;
typedef __attribute__((ext_vector_type(8)))  float    v8f;

// WMMA f32 = f16 x f16 + f32, 16x16x32, wave32
__device__ __forceinline__ v8f wmma_f16(v16h a, v16h b, v8f c) {
  return __builtin_amdgcn_wmma_f32_16x16x32_f16(false, a, false, b, (short)0, c,
                                                false, false);
}

#if USE_ASYNC_CP
// GLOBAL_LOAD_ASYNC_TO_LDS_B128 (ISA 15.18.3 op 98, GV mode):
// per-lane copy of 16B from global memory to LDS, tracked by ASYNCcnt.
// VDST holds the LDS byte offset (= low 32 bits of the generic LDS VA,
// ISA 10.2 aperture truncation); VADDR is the 64-bit global address.
__device__ __forceinline__ void async_cp16B(const f16* g, f16* l) {
  unsigned lds_off = (unsigned)(uintptr_t)l;
  asm volatile("global_load_async_to_lds_b128 %0, %1, off"
               :
               : "v"(lds_off), "v"(g)
               : "memory");
}
__device__ __forceinline__ void wait_async0() {
  asm volatile("s_wait_asynccnt 0x0" ::: "memory");
}
#else
__device__ __forceinline__ void async_cp16B(const f16* g, f16* l) {
  *(v8h*)l = *(const v8h*)g;
}
__device__ __forceinline__ void wait_async0() {}
#endif

// Load a 16x32 f16 A/B fragment from LDS (row-major, stride ld halves).
// ISA 7.12.2 16-bit layout: lane half t holds K = t + 8*(lane>=16) + 8*(t>=8).
__device__ __forceinline__ v16h load_frag(const f16* base, int row_base,
                                          int k_base, int ld, int lane) {
  const f16* p = base + (size_t)(row_base + (lane & 15)) * ld;
  int k0 = k_base + ((lane >> 4) << 3);
  v8h lo = *(const v8h*)(p + k0);
  v8h hi = *(const v8h*)(p + k0 + 16);
  v16h r;
#pragma unroll
  for (int t = 0; t < 8; ++t) { r[t] = lo[t]; r[t + 8] = hi[t]; }
  return r;
}

// ---------------- f32 -> f16 convert ----------------
__global__ void cvt_f32_to_f16(const float* __restrict__ in, f16* __restrict__ out,
                               size_t n) {
  size_t i = (size_t)blockIdx.x * blockDim.x + threadIdx.x;
  size_t stride = (size_t)gridDim.x * blockDim.x;
  for (; i < n; i += stride) out[i] = (f16)in[i];
}

// ---------------- GEMM: C[M,N] = A[M,K] * W[N,K]^T + bias, epilogue MODE ------
// 128x128x32 tiles, 8 waves (4x2), double-buffered LDS with async fills.
enum { OUT_F32 = 0, OUT_F16 = 1, OUT_F16_KSCALE = 2, OUT_F32_GELU = 3,
       OUT_F32_RESID = 4 };

template <int MODE>
__global__ __launch_bounds__(256) void gemm_f16(
    const f16* __restrict__ A, const f16* __restrict__ W,
    const float* __restrict__ bias, const float* __restrict__ resid,
    float* __restrict__ outf, f16* __restrict__ outh, int M, int N, int K) {
  __shared__ f16 As[2][128 * 32];
  __shared__ f16 Bs[2][128 * 32];
  const int tid = threadIdx.x;
  const int lane = tid & 31;
  const int wave = tid >> 5;   // 0..7
  const int wm = wave >> 1;    // 0..3 : 32 rows each
  const int wn = wave & 1;     // 0..1 : 64 cols each
  const int bm = blockIdx.y * 128;
  const int bn = blockIdx.x * 128;

  v8f acc[2][4];
#pragma unroll
  for (int i = 0; i < 2; ++i)
#pragma unroll
    for (int j = 0; j < 4; ++j) acc[i][j] = {};

  const int r = tid >> 1;          // 0..127
  const int c0 = (tid & 1) * 16;   // 0 or 16

  auto issue_stage = [&](int stage, int k0) {
    const f16* ga = A + (size_t)(bm + r) * K + k0 + c0;
    const f16* gb = W + (size_t)(bn + r) * K + k0 + c0;
    f16* la = &As[stage][r * 32 + c0];
    f16* lb = &Bs[stage][r * 32 + c0];
    async_cp16B(ga, la);
    async_cp16B(ga + 8, la + 8);
    async_cp16B(gb, lb);
    async_cp16B(gb + 8, lb + 8);
  };

  auto compute_stage = [&](int stage) {
    v16h af[2], bf[4];
#pragma unroll
    for (int mt = 0; mt < 2; ++mt)
      af[mt] = load_frag(As[stage], wm * 32 + mt * 16, 0, 32, lane);
#pragma unroll
    for (int nt = 0; nt < 4; ++nt)
      bf[nt] = load_frag(Bs[stage], wn * 64 + nt * 16, 0, 32, lane);
#pragma unroll
    for (int mt = 0; mt < 2; ++mt)
#pragma unroll
      for (int nt = 0; nt < 4; ++nt)
        acc[mt][nt] = wmma_f16(af[mt], bf[nt], acc[mt][nt]);
  };

  issue_stage(0, 0);
  for (int k0 = 0; k0 < K; k0 += 64) {
    wait_async0();
    __syncthreads();
    if (k0 + 32 < K) issue_stage(1, k0 + 32);
    compute_stage(0);
    wait_async0();
    __syncthreads();
    if (k0 + 64 < K) issue_stage(0, k0 + 64);
    compute_stage(1);
  }

#pragma unroll
  for (int mt = 0; mt < 2; ++mt)
#pragma unroll
    for (int nt = 0; nt < 4; ++nt)
#pragma unroll
      for (int v = 0; v < 8; ++v) {
        int row = bm + wm * 32 + mt * 16 + ((lane >> 4) << 3) + v;
        int col = bn + wn * 64 + nt * 16 + (lane & 15);
        float val = acc[mt][nt][v] + bias[col];
        size_t o = (size_t)row * N + col;
        if (MODE == OUT_F32) outf[o] = val;
        if (MODE == OUT_F16) outh[o] = (f16)val;
        if (MODE == OUT_F16_KSCALE) outh[o] = (f16)(val * 0.125f);  // 64^-0.5
        if (MODE == OUT_F32_GELU)
          outf[o] = 0.5f * val * (1.0f + erff(val * 0.70710678118f));
        if (MODE == OUT_F32_RESID) outf[o] = val + resid[o];
      }
}

// ---------------- fused retention: o = ((QK^T+DM)V)/den, head-LN, SiLU gate ---
__global__ __launch_bounds__(128) void retention_kernel(
    const f16* __restrict__ Qh, const f16* __restrict__ Kh,
    const f16* __restrict__ Vh, const float* __restrict__ Gf,
    f16* __restrict__ attn_h) {
  __shared__ f16 Qs[64 * 64];
  __shared__ f16 Ks[64 * 64];
  __shared__ f16 Vts[64 * 64];   // transposed: [d][j]
  __shared__ f16 Ps[64 * 64];
  __shared__ float rowsum[64];
  __shared__ float obuf[64 * 64];

  const int tid = threadIdx.x;
  const int lane = tid & 31;
  const int wave = tid >> 5;             // 0..3, 16 rows each
  const int bh = blockIdx.y;
  const int b = bh / H, h = bh % H;
  const int i0 = blockIdx.x * 64;

  const float gamma = 1.0f - exp2f(-5.0f - (float)h);
  const float logg = logf(gamma);
  const float inv1mg = 1.0f / (1.0f - gamma);

  // stage Q block (64 rows x 64)
  {
    int r = tid >> 1, c0 = (tid & 1) * 32;
    const f16* p = Qh + ((size_t)(b * S + i0 + r)) * D + h * KD + c0;
#pragma unroll
    for (int t = 0; t < 4; ++t)
      *(v8h*)(&Qs[r * 64 + c0 + t * 8]) = *(const v8h*)(p + t * 8);
  }
  if (tid < 64) rowsum[tid] = 0.0f;

  v8f oacc[4];
#pragma unroll
  for (int nt = 0; nt < 4; ++nt) oacc[nt] = {};
  __syncthreads();

  v16h aq0 = load_frag(Qs, wave * 16, 0, 64, lane);
  v16h aq1 = load_frag(Qs, wave * 16, 32, 64, lane);

  for (int j0 = 0; j0 < S; j0 += 64) {
    // stage K (row-major) and V (transposed) tiles
    {
      int r = tid >> 1, c0 = (tid & 1) * 32;
      const f16* kp = Kh + ((size_t)(b * S + j0 + r)) * D + h * KD + c0;
      const f16* vp = Vh + ((size_t)(b * S + j0 + r)) * D + h * KD + c0;
#pragma unroll
      for (int t = 0; t < 4; ++t)
        *(v8h*)(&Ks[r * 64 + c0 + t * 8]) = *(const v8h*)(kp + t * 8);
      v8h vt[4];
#pragma unroll
      for (int t = 0; t < 4; ++t) vt[t] = *(const v8h*)(vp + t * 8);
#pragma unroll
      for (int t = 0; t < 4; ++t)
#pragma unroll
        for (int c = 0; c < 8; ++c) Vts[(c0 + t * 8 + c) * 64 + r] = vt[t][c];
    }
    __syncthreads();

    // P = Q K^T + decay bias, store as f16 in LDS
#pragma unroll
    for (int nt = 0; nt < 4; ++nt) {
      v8f p = {};
      v16h bk0 = load_frag(Ks, nt * 16, 0, 64, lane);
      v16h bk1 = load_frag(Ks, nt * 16, 32, 64, lane);
      p = wmma_f16(aq0, bk0, p);
      p = wmma_f16(aq1, bk1, p);
      int jl = nt * 16 + (lane & 15);
      int jg = j0 + jl;
#pragma unroll
      for (int v = 0; v < 8; ++v) {
        int il = wave * 16 + ((lane >> 4) << 3) + v;
        int ig = i0 + il;
        float val = p[v];
        if (ig >= jg) {
          float rs = (1.0f - expf((float)(ig + 1) * logg)) * inv1mg;
          val += expf((float)(ig - jg) * logg) * rsqrtf(rs);
        }
        Ps[il * 64 + jl] = (f16)val;
      }
    }
    __syncthreads();

    if (tid < 64) {  // accumulate row sums for the detached L1 normalizer
      float s = 0.0f;
      for (int j = 0; j < 64; ++j) s += (float)Ps[tid * 64 + j];
      rowsum[tid] += s;
    }

    // o += P V
    v16h ap0 = load_frag(Ps, wave * 16, 0, 64, lane);
    v16h ap1 = load_frag(Ps, wave * 16, 32, 64, lane);
#pragma unroll
    for (int nt = 0; nt < 4; ++nt) {
      v16h bv0 = load_frag(Vts, nt * 16, 0, 64, lane);
      v16h bv1 = load_frag(Vts, nt * 16, 32, 64, lane);
      oacc[nt] = wmma_f16(ap0, bv0, oacc[nt]);
      oacc[nt] = wmma_f16(ap1, bv1, oacc[nt]);
    }
    __syncthreads();
  }

  // apply den, stash to LDS for per-head LayerNorm
#pragma unroll
  for (int nt = 0; nt < 4; ++nt)
#pragma unroll
    for (int v = 0; v < 8; ++v) {
      int il = wave * 16 + ((lane >> 4) << 3) + v;
      int dl = nt * 16 + (lane & 15);
      float den = fmaxf(fabsf(rowsum[il]), 1.0f);
      obuf[il * 64 + dl] = oacc[nt][v] / den;
    }
  __syncthreads();

  if (tid < 64) {  // head-LN (eps 1e-6, no affine) + SiLU(g) gate, store f16
    float m = 0.0f;
    for (int d = 0; d < 64; ++d) m += obuf[tid * 64 + d];
    m *= (1.0f / 64.0f);
    float var = 0.0f;
    for (int d = 0; d < 64; ++d) {
      float t = obuf[tid * 64 + d] - m;
      var += t * t;
    }
    float rstd = rsqrtf(var * (1.0f / 64.0f) + 1e-6f);
    const float* gp = Gf + ((size_t)(b * S + i0 + tid)) * D + h * KD;
    f16* op = attn_h + ((size_t)(b * S + i0 + tid)) * D + h * KD;
    for (int d = 0; d < 64; ++d) {
      float g = gp[d];
      float sg = g / (1.0f + expf(-g));
      op[d] = (f16)(sg * (obuf[tid * 64 + d] - m) * rstd);
    }
  }
}

// ---------------- row LayerNorm (f32 in; optional f32 and/or f16 out) --------
__global__ __launch_bounds__(256) void layernorm_row(
    const float* __restrict__ in, const float* __restrict__ g,
    const float* __restrict__ bta, float eps, int rowlen,
    float* __restrict__ outf, f16* __restrict__ outh) {
  __shared__ float red[256];
  const int tid = threadIdx.x;
  const float* x = in + (size_t)blockIdx.x * rowlen;
  float s = 0.0f;
  for (int i = tid; i < rowlen; i += 256) s += x[i];
  red[tid] = s;
  __syncthreads();
  for (int off = 128; off > 0; off >>= 1) {
    if (tid < off) red[tid] += red[tid + off];
    __syncthreads();
  }
  float mu = red[0] / (float)rowlen;
  __syncthreads();
  float v = 0.0f;
  for (int i = tid; i < rowlen; i += 256) {
    float t = x[i] - mu;
    v += t * t;
  }
  red[tid] = v;
  __syncthreads();
  for (int off = 128; off > 0; off >>= 1) {
    if (tid < off) red[tid] += red[tid + off];
    __syncthreads();
  }
  float rstd = rsqrtf(red[0] / (float)rowlen + eps);
  for (int i = tid; i < rowlen; i += 256) {
    float y = (x[i] - mu) * rstd * g[i] + bta[i];
    if (outf) outf[(size_t)blockIdx.x * rowlen + i] = y;
    if (outh) outh[(size_t)blockIdx.x * rowlen + i] = (f16)y;
  }
}

// ---------------- host orchestration ----------------
extern "C" void kernel_launch(void* const* d_in, const int* in_sizes, int n_in,
                              void* d_out, int out_size, void* d_ws,
                              size_t ws_size, hipStream_t stream) {
  (void)in_sizes; (void)n_in; (void)out_size; (void)ws_size;
  const float* src   = (const float*)d_in[0];
  const float* wq    = (const float*)d_in[3];
  const float* bq    = (const float*)d_in[4];
  const float* wk    = (const float*)d_in[5];
  const float* bk    = (const float*)d_in[6];
  const float* wv    = (const float*)d_in[7];
  const float* bv    = (const float*)d_in[8];
  const float* wg    = (const float*)d_in[9];
  const float* bg    = (const float*)d_in[10];
  const float* wo    = (const float*)d_in[11];
  const float* bo    = (const float*)d_in[12];
  const float* fc1w  = (const float*)d_in[13];
  const float* fc1b  = (const float*)d_in[14];
  const float* fc2w  = (const float*)d_in[15];
  const float* fc2b  = (const float*)d_in[16];
  const float* lnfg  = (const float*)d_in[17];
  const float* lnfb  = (const float*)d_in[18];
  const float* n0g   = (const float*)d_in[19];
  const float* n0b   = (const float*)d_in[20];
  const float* n1g   = (const float*)d_in[21];
  const float* n1b   = (const float*)d_in[22];
  float* out = (float*)d_out;

  char* w = (char*)d_ws;
  auto alloc = [&](size_t bytes) -> void* {
    void* p = (void*)w;
    w += (bytes + 255) & ~(size_t)255;
    return p;
  };
  f16* Xh    = (f16*)alloc((size_t)NTOK * D * 2);
  f16* wq_h  = (f16*)alloc((size_t)D * D * 2);
  f16* wk_h  = (f16*)alloc((size_t)D * D * 2);
  f16* wv_h  = (f16*)alloc((size_t)D * D * 2);
  f16* wg_h  = (f16*)alloc((size_t)D * D * 2);
  f16* wo_h  = (f16*)alloc((size_t)D * D * 2);
  f16* fc1_h = (f16*)alloc((size_t)FFN * D * 2);
  f16* fc2_h = (f16*)alloc((size_t)D * FFN * 2);
  f16* Qh    = (f16*)alloc((size_t)NTOK * D * 2);
  f16* Kh    = (f16*)alloc((size_t)NTOK * D * 2);
  f16* Vh    = (f16*)alloc((size_t)NTOK * D * 2);
  float* Gf  = (float*)alloc((size_t)NTOK * D * 4);
  f16* Ah    = (f16*)alloc((size_t)NTOK * D * 2);   // gated attention out, f16
  float* Y0  = (float*)alloc((size_t)NTOK * D * 4); // src + attn
  float* X0  = (float*)alloc((size_t)NTOK * D * 4); // LN(Y0) f32
  f16* X0h   = (f16*)alloc((size_t)NTOK * D * 2);
  float* H1  = (float*)alloc((size_t)NTOK * FFN * 4); // gelu(fc1)
  f16* H2h   = (f16*)alloc((size_t)NTOK * FFN * 2);   // subln f16
  float* Y1  = (float*)alloc((size_t)NTOK * D * 4);   // x + fc2

  // ---- converts
  cvt_f32_to_f16<<<2048, 256, 0, stream>>>(src, Xh, (size_t)NTOK * D);
  cvt_f32_to_f16<<<512, 256, 0, stream>>>(wq, wq_h, (size_t)D * D);
  cvt_f32_to_f16<<<512, 256, 0, stream>>>(wk, wk_h, (size_t)D * D);
  cvt_f32_to_f16<<<512, 256, 0, stream>>>(wv, wv_h, (size_t)D * D);
  cvt_f32_to_f16<<<512, 256, 0, stream>>>(wg, wg_h, (size_t)D * D);
  cvt_f32_to_f16<<<512, 256, 0, stream>>>(wo, wo_h, (size_t)D * D);
  cvt_f32_to_f16<<<1024, 256, 0, stream>>>(fc1w, fc1_h, (size_t)FFN * D);
  cvt_f32_to_f16<<<1024, 256, 0, stream>>>(fc2w, fc2_h, (size_t)D * FFN);

  // ---- projections
  dim3 blk(256);
  dim3 gridP(D / 128, NTOK / 128);
  gemm_f16<OUT_F16><<<gridP, blk, 0, stream>>>(Xh, wq_h, bq, nullptr, nullptr,
                                               Qh, NTOK, D, D);
  gemm_f16<OUT_F16_KSCALE><<<gridP, blk, 0, stream>>>(Xh, wk_h, bk, nullptr,
                                                      nullptr, Kh, NTOK, D, D);
  gemm_f16<OUT_F16><<<gridP, blk, 0, stream>>>(Xh, wv_h, bv, nullptr, nullptr,
                                               Vh, NTOK, D, D);
  gemm_f16<OUT_F32><<<gridP, blk, 0, stream>>>(Xh, wg_h, bg, nullptr, Gf,
                                               nullptr, NTOK, D, D);

  // ---- fused retention + head-LN + SiLU gate
  retention_kernel<<<dim3(S / 64, B * H), dim3(128), 0, stream>>>(Qh, Kh, Vh,
                                                                  Gf, Ah);

  // ---- output projection + residual
  gemm_f16<OUT_F32_RESID><<<gridP, blk, 0, stream>>>(Ah, wo_h, bo, src, Y0,
                                                     nullptr, NTOK, D, D);
  layernorm_row<<<NTOK, 256, 0, stream>>>(Y0, n0g, n0b, 1e-5f, D, X0, X0h);

  // ---- FFN with subln
  dim3 gridF1(FFN / 128, NTOK / 128);
  gemm_f16<OUT_F32_GELU><<<gridF1, blk, 0, stream>>>(X0h, fc1_h, fc1b, nullptr,
                                                     H1, nullptr, NTOK, FFN, D);
  layernorm_row<<<NTOK, 256, 0, stream>>>(H1, lnfg, lnfb, 1e-6f, FFN, nullptr,
                                          H2h);
  gemm_f16<OUT_F32_RESID><<<gridP, blk, 0, stream>>>(H2h, fc2_h, fc2b, X0, Y1,
                                                     nullptr, NTOK, D, FFN);

  // ---- final LN to output
  layernorm_row<<<NTOK, 256, 0, stream>>>(Y1, n1g, n1b, 1e-5f, D, out, nullptr);
}
